// UniGATConv_50749333569738
// MI455X (gfx1250) — matrix operand order repaired and checked by reading, not compiled, not run on hardware
//
#include <hip/hip_runtime.h>
#include <hip/hip_bf16.h>
#include <math.h>

typedef __attribute__((ext_vector_type(16))) _Float16 v16h;
typedef __attribute__((ext_vector_type(8)))  _Float16 v8h;
typedef __attribute__((ext_vector_type(8)))  float    v8f;

#define N_NODES   50000
#define N_EDGES   50000
#define NNZ       800000
#define IN_C      128
#define HEADS     8
#define OUT_C     16
#define HC        128           // HEADS*OUT_C
#define NEG_SLOPE 0.2f
#define ROWS_PER_BLK 64
#define N_BLKS    ((N_NODES + ROWS_PER_BLK - 1) / ROWS_PER_BLK)   // 782
#define N_PAD     (N_BLKS * ROWS_PER_BLK)                         // 50048 (padded rows)

// ---------------------------------------------------------------------------
// Utility fill kernel (grid-stride)
// ---------------------------------------------------------------------------
__global__ void fill_f32(float* __restrict__ p, float v, long n) {
    long i = (long)blockIdx.x * blockDim.x + threadIdx.x;
    long stride = (long)gridDim.x * blockDim.x;
    for (; i < n; i += stride) p[i] = v;
}

// ---------------------------------------------------------------------------
// Stage 1: Xh = X @ W via v_wmma_f32_16x16x32_f16
// Block: 256 threads (8 waves), 64 rows (4 row-tiles). Wave w = col tile w.
// W staged in LDS in fragment order (each lane's B frag = 32 contiguous B).
// Xh is padded to N_PAD rows so D stores need no tail guard (EXEC stays full,
// stores form a clean clause).
// ---------------------------------------------------------------------------
__global__ __launch_bounds__(256)
void gemm_wmma_kernel(const float* __restrict__ X, const float* __restrict__ W,
                      float* __restrict__ Xh) {
    // sWf[((ks*2+kg)*128 + n)*16 + j] : fragment-order W, 32 KB
    __shared__ _Float16 sWf[IN_C * HC];
    // sA[m][k] for 64 rows, 16 KB
    __shared__ _Float16 sA[ROWS_PER_BLK * IN_C];

    const int tid = threadIdx.x;
    const int rowbase = blockIdx.x * ROWS_PER_BLK;

    // Swizzle W into fragment order (coalesced global reads).
    for (int j = tid; j < IN_C * HC; j += 256) {
        int k = j >> 7, n = j & 127;
        int ks = k >> 5, kg = (k >> 4) & 1, jj = k & 15;
        sWf[((((ks << 1) + kg) << 7) + n) * 16 + jj] = (_Float16)W[j];
    }
    // Stage the 64-row A tile (clamp reads for the final partial block).
    for (int j = tid; j < ROWS_PER_BLK * IN_C; j += 256) {
        int m = j >> 7, k = j & 127;
        int row = rowbase + m;
        if (row >= N_NODES) row = N_NODES - 1;
        sA[j] = (_Float16)X[(size_t)row * IN_C + k];
    }
    __syncthreads();

    const int lane = tid & 31;
    const int wave = tid >> 5;         // 0..7 -> output column tile
    const int ml   = lane & 15;        // M (A) / N (B,C,D)
    const int kg   = lane >> 4;        // lane-half selector
    const int ncol = wave * 16 + ml;   // global output column

    // Load all 4 B fragments once (each: 32 contiguous bytes in LDS).
    v16h bfrag[4];
    #pragma unroll
    for (int ks = 0; ks < 4; ++ks) {
        const _Float16* bp = &sWf[((((ks << 1) + kg) << 7) + ncol) * 16];
        bfrag[ks] = *(const v16h*)bp;
    }

    // 4 row tiles of 16 rows each.
    #pragma unroll
    for (int rt = 0; rt < 4; ++rt) {
        const int rowb = rt * 16;
        v8f acc = {};
        #pragma unroll
        for (int ks = 0; ks < 4; ++ks) {
            // A fragment: lane ml holds row rowb+ml; one contiguous 32B run:
            // j=0..15 -> k = ks*32 + 16*kg + j
            const _Float16* ap = &sA[(rowb + ml) * IN_C + ks * 32 + 16 * kg];
            v8h alo = *(const v8h*)ap;
            v8h ahi = *(const v8h*)(ap + 8);
            v16h a;
            #pragma unroll
            for (int j = 0; j < 8; ++j) { a[j] = alo[j]; a[j + 8] = ahi[j]; }

            acc = __builtin_amdgcn_wmma_f32_16x16x32_f16(
                false, a, false, bfrag[ks], (short)0, acc, false, false);
        }
        // C/D layout: VGPR r -> row m = r + 8*kg, col = lane&15.
        // Xh is padded to N_PAD rows: no tail guard needed.
        float* dst = Xh + (size_t)(rowbase + rowb + 8 * kg) * HC + ncol;
        #pragma unroll
        for (int r = 0; r < 8; ++r) {
            dst[(size_t)r * HC] = acc[r];
        }
    }
}

// ---------------------------------------------------------------------------
// Stage 2: scatter node feats to hyperedges (sums + counts)
// ---------------------------------------------------------------------------
__global__ void scatter_edge_kernel(const float* __restrict__ Xh,
                                    const int* __restrict__ vertex,
                                    const int* __restrict__ edges,
                                    float* __restrict__ sums,
                                    float* __restrict__ cnt) {
    long idx = (long)blockIdx.x * blockDim.x + threadIdx.x;
    if (idx >= (long)NNZ * HC) return;
    int inc = (int)(idx >> 7);
    int c   = (int)(idx & 127);
    int v = vertex[inc];
    int e = edges[inc];
    atomicAdd(&sums[(size_t)e * HC + c], Xh[(size_t)v * HC + c]);
    if (c == 0) atomicAdd(&cnt[e], 1.0f);
}

// ---------------------------------------------------------------------------
// Stage 3: finalize Xe = sums/cnt (in place) and alpha_e = sum_c Xe*att_e
// ---------------------------------------------------------------------------
__global__ __launch_bounds__(128)
void edge_finalize_kernel(float* __restrict__ Xe,            // in: sums, out: Xe
                          const float* __restrict__ cnt,
                          const float* __restrict__ att_e,   // 128 floats
                          float* __restrict__ alpha_e) {     // [E, 8]
    int e = blockIdx.x;
    int c = threadIdx.x;
    float inv = 1.0f / fmaxf(cnt[e], 1.0f);
    float xe = Xe[(size_t)e * HC + c] * inv;
    Xe[(size_t)e * HC + c] = xe;
    float prod = xe * att_e[c];
    // reduce across the 16 lanes of each head (xor over low 4 bits)
    #pragma unroll
    for (int off = 8; off >= 1; off >>= 1)
        prod += __shfl_xor(prod, off, 32);
    if ((c & 15) == 0) alpha_e[(size_t)e * HEADS + (c >> 4)] = prod;
}

// ---------------------------------------------------------------------------
// Float atomic max handling negatives (int max / uint min trick)
// ---------------------------------------------------------------------------
__device__ __forceinline__ void atomicMaxFloat(float* addr, float val) {
    if (val >= 0.0f)
        atomicMax((int*)addr, __float_as_int(val));
    else
        atomicMin((unsigned int*)addr, __float_as_uint(val));
}

// Stage 4a: segment max of leaky-relu logits keyed by vertex
__global__ void seg_max_kernel(const float* __restrict__ alpha_e,
                               const int* __restrict__ vertex,
                               const int* __restrict__ edges,
                               float* __restrict__ amax) {
    long idx = (long)blockIdx.x * blockDim.x + threadIdx.x;
    if (idx >= (long)NNZ * HEADS) return;
    int inc = (int)(idx >> 3);
    int h   = (int)(idx & 7);
    float a = alpha_e[(size_t)edges[inc] * HEADS + h];
    a = (a >= 0.0f) ? a : NEG_SLOPE * a;
    atomicMaxFloat(&amax[(size_t)vertex[inc] * HEADS + h], a);
}

// Stage 4b: denom = segment_sum(exp(a - amax[vertex]))
__global__ void seg_denom_kernel(const float* __restrict__ alpha_e,
                                 const int* __restrict__ vertex,
                                 const int* __restrict__ edges,
                                 const float* __restrict__ amax,
                                 float* __restrict__ denom) {
    long idx = (long)blockIdx.x * blockDim.x + threadIdx.x;
    if (idx >= (long)NNZ * HEADS) return;
    int inc = (int)(idx >> 3);
    int h   = (int)(idx & 7);
    int v = vertex[inc];
    float a = alpha_e[(size_t)edges[inc] * HEADS + h];
    a = (a >= 0.0f) ? a : NEG_SLOPE * a;
    float ex = expf(a - amax[(size_t)v * HEADS + h]);
    atomicAdd(&denom[(size_t)v * HEADS + h], ex);
}

// ---------------------------------------------------------------------------
// Stage 5: weighted scatter of edge feats back to vertices (into d_out)
// ---------------------------------------------------------------------------
__global__ void scatter_vertex_kernel(const float* __restrict__ Xe,
                                      const float* __restrict__ alpha_e,
                                      const int* __restrict__ vertex,
                                      const int* __restrict__ edges,
                                      const float* __restrict__ amax,
                                      const float* __restrict__ denom,
                                      float* __restrict__ out) {
    long idx = (long)blockIdx.x * blockDim.x + threadIdx.x;
    if (idx >= (long)NNZ * HC) return;
    int inc = (int)(idx >> 7);
    int c   = (int)(idx & 127);
    int h   = c >> 4;
    int v = vertex[inc];
    int e = edges[inc];
    float a = alpha_e[(size_t)e * HEADS + h];
    a = (a >= 0.0f) ? a : NEG_SLOPE * a;
    float ex = expf(a - amax[(size_t)v * HEADS + h]);
    float alpha = ex / (denom[(size_t)v * HEADS + h] + 1e-16f);
    atomicAdd(&out[(size_t)v * HC + c], Xe[(size_t)e * HC + c] * alpha);
}

// ---------------------------------------------------------------------------
// Stage 6: row L2 normalize (in place on d_out). One wave per row.
// ---------------------------------------------------------------------------
__global__ __launch_bounds__(256)
void normalize_kernel(float* __restrict__ out) {
    int row  = blockIdx.x * 8 + (threadIdx.x >> 5);
    int lane = threadIdx.x & 31;
    if (row >= N_NODES) return;
    float4* p = (float4*)(out + (size_t)row * HC);
    float4 v = p[lane];                     // 32 lanes x 4 = 128
    float ss = v.x * v.x + v.y * v.y + v.z * v.z + v.w * v.w;
    #pragma unroll
    for (int off = 16; off >= 1; off >>= 1)
        ss += __shfl_xor(ss, off, 32);
    float rn = sqrtf(ss);
    float s = (rn > 0.0f) ? (1.0f / rn) : 0.0f;
    v.x *= s; v.y *= s; v.z *= s; v.w *= s;
    p[lane] = v;
}

// ---------------------------------------------------------------------------
// Launcher
// ---------------------------------------------------------------------------
static inline size_t align_up(size_t x) { return (x + 255) & ~(size_t)255; }

extern "C" void kernel_launch(void* const* d_in, const int* in_sizes, int n_in,
                              void* d_out, int out_size, void* d_ws, size_t ws_size,
                              hipStream_t stream) {
    const float* X      = (const float*)d_in[0];  // [N, 128]
    const float* W      = (const float*)d_in[1];  // [128, 128]
    const float* att_e  = (const float*)d_in[2];  // [1, 8, 16] -> 128
    const int*   vertex = (const int*)d_in[3];    // [NNZ]
    const int*   edges  = (const int*)d_in[4];    // [NNZ]
    float* out = (float*)d_out;                   // [N, 128]

    // Workspace partition (Xh padded to N_PAD rows for guard-free stores)
    char* w = (char*)d_ws;
    float* Xh      = (float*)w; w += align_up((size_t)N_PAD * HC * 4);     // ~25.6 MB
    float* Xe      = (float*)w; w += align_up((size_t)N_EDGES * HC * 4);   // 25.6 MB (sums -> Xe)
    float* cnt     = (float*)w; w += align_up((size_t)N_EDGES * 4);
    float* alpha_e = (float*)w; w += align_up((size_t)N_EDGES * HEADS * 4);
    float* amax    = (float*)w; w += align_up((size_t)N_NODES * HEADS * 4);
    float* denom   = (float*)w; w += align_up((size_t)N_NODES * HEADS * 4);

    // ---- init accumulators (graph-replay safe) ----
    {
        long n;
        n = (long)N_EDGES * HC;
        fill_f32<<<(unsigned)((n + 255) / 256), 256, 0, stream>>>(Xe, 0.0f, n);
        n = (long)N_EDGES;
        fill_f32<<<(unsigned)((n + 255) / 256), 256, 0, stream>>>(cnt, 0.0f, n);
        n = (long)N_NODES * HEADS;
        fill_f32<<<(unsigned)((n + 255) / 256), 256, 0, stream>>>(amax, -INFINITY, n);
        fill_f32<<<(unsigned)((n + 255) / 256), 256, 0, stream>>>(denom, 0.0f, n);
        n = (long)N_NODES * HC;
        fill_f32<<<(unsigned)((n + 255) / 256), 256, 0, stream>>>(out, 0.0f, n);
    }

    // ---- Stage 1: WMMA GEMM ----
    gemm_wmma_kernel<<<N_BLKS, 256, 0, stream>>>(X, W, Xh);

    // ---- Stage 2: scatter to edges ----
    {
        long n = (long)NNZ * HC;
        scatter_edge_kernel<<<(unsigned)((n + 255) / 256), 256, 0, stream>>>(
            Xh, vertex, edges, Xe, cnt);
    }

    // ---- Stage 3: edge mean + attention logits ----
    edge_finalize_kernel<<<N_EDGES, 128, 0, stream>>>(Xe, cnt, att_e, alpha_e);

    // ---- Stage 4: segment softmax (max then denom) ----
    {
        long n = (long)NNZ * HEADS;
        seg_max_kernel<<<(unsigned)((n + 255) / 256), 256, 0, stream>>>(
            alpha_e, vertex, edges, amax);
        seg_denom_kernel<<<(unsigned)((n + 255) / 256), 256, 0, stream>>>(
            alpha_e, vertex, edges, amax, denom);
    }

    // ---- Stage 5: weighted scatter back to vertices ----
    {
        long n = (long)NNZ * HC;
        scatter_vertex_kernel<<<(unsigned)((n + 255) / 256), 256, 0, stream>>>(
            Xe, alpha_e, vertex, edges, amax, denom, out);
    }

    // ---- Stage 6: row L2 normalize ----
    normalize_kernel<<<(N_NODES + 7) / 8, 256, 0, stream>>>(out);
}
